// Model_13778255085586
// MI455X (gfx1250) — matrix operand order repaired and checked by reading, not compiled, not run on hardware
//
#include <hip/hip_runtime.h>
#include <hip/hip_bf16.h>

typedef __attribute__((ext_vector_type(16))) __bf16 v16bf;
typedef __attribute__((ext_vector_type(8)))  float  v8f;

static constexpr int Bb = 256;   // batch
static constexpr int Tt = 256;   // time
static constexpr int Cc = 64;    // features
static constexpr int Hh = 512;   // hidden

__device__ __forceinline__ v8f wmma_bf16(v16bf a, v16bf b, v8f c) {
  return __builtin_amdgcn_wmma_f32_16x16x32_bf16(false, a, false, b, (short)0, c, false, false);
}

// A-fragment (16x32 bf16) LDS address for element (row m, k)
__device__ __forceinline__ int afrag_addr(int m, int k) {
  int f  = k >> 5;
  int kk = k & 31;
  int ln = ((kk & 15) >> 3) * 16 + m;     // lane
  int e  = (kk >> 4) * 8 + (kk & 7);      // element within lane
  return f * 512 + ln * 16 + e;
}

__device__ __forceinline__ v16bf load_afrag(const __bf16* lds, int f, int lane) {
  return *(const v16bf*)(lds + f * 512 + lane * 16);
}

// Packed B-fragments in global: frag (kt*Ntiles + nt), 32 lanes x 16 bf16 contiguous
__device__ __forceinline__ v16bf load_bfrag(const __bf16* base, int kt, int nt, int Ntiles, int lane) {
  return *(const v16bf*)(base + (size_t)(kt * Ntiles + nt) * 512 + lane * 16);
}

__device__ __forceinline__ float sigm(float x) { return 1.f / (1.f + __expf(-x)); }

// ---------------- prep: masks / values / deltas ----------------
__global__ __launch_bounds__(256) void prep_kernel(const float* __restrict__ data,
                                                   float* __restrict__ values,
                                                   float* __restrict__ masks,
                                                   float* __restrict__ deltas) {
  int i = blockIdx.x * blockDim.x + threadIdx.x;
  if (i >= Bb * Cc) return;
  int b = i / Cc, c = i % Cc;
  float decay = 1.f;
  float mprev = 1.f;
  for (int t = 0; t < Tt; ++t) {
    size_t idx = ((size_t)b * Tt + t) * Cc + c;
    float v = data[idx];
    bool nn = (v != v);
    float m = nn ? 0.f : 1.f;
    masks[idx]  = m;
    values[idx] = nn ? 0.f : v;
    float d;
    if (t == 0)      d = 0.f;
    else if (t == 1) d = 1.f;
    else { decay = (mprev == 1.f) ? 1.f : decay + 1.f; d = decay; }
    deltas[idx] = d;
    mprev = m;
  }
}

// ---------------- pack W (row-major [N][K]) -> WMMA B-fragments bf16 ----------------
__global__ __launch_bounds__(256) void pack_kernel(const float* __restrict__ W, __bf16* __restrict__ dst,
                                                   int N, int K, int zero_diag) {
  int total = N * K;
  int Nt = N >> 4;
  for (int i = blockIdx.x * blockDim.x + threadIdx.x; i < total; i += gridDim.x * blockDim.x) {
    int f = i >> 9, r = i & 511;
    int L = r >> 4, e = r & 15;
    int kt = f / Nt, nt = f % Nt;
    int k = kt * 32 + (e >> 3) * 16 + (L >> 4) * 8 + (e & 7);
    int n = nt * 16 + (L & 15);
    float v = W[(size_t)n * K + k];
    if (zero_diag && n == k) v = 0.f;
    dst[i] = (__bf16)v;
  }
}

// ---------------- persistent scan: one block owns a 16-row tile for all T steps ----------------
// grid = Bb/16 = 16 blocks, block = 256 threads (8 waves)
__global__ __launch_bounds__(256) void scan_kernel(
    const float* __restrict__ values, const float* __restrict__ masks, const float* __restrict__ deltas,
    const __bf16* __restrict__ Wgh_p,   const float* __restrict__ b_gh,
    const __bf16* __restrict__ Whist_p, const float* __restrict__ b_hist,
    const __bf16* __restrict__ Wfeat_p, const float* __restrict__ b_feat,
    const __bf16* __restrict__ Wcomb_p, const float* __restrict__ b_comb,
    const float* __restrict__ Wgx,      const float* __restrict__ b_gx,
    const __bf16* __restrict__ Wih_p,   const float* __restrict__ b_ih,
    const __bf16* __restrict__ Whh_p,   const float* __restrict__ b_hh,
    float* __restrict__ out) {

  __shared__ __bf16 lds_h   [16 * 512];  // hidden state, row-major [m][k], bf16 (16KB)
  __shared__ __bf16 lds_hdec[16 * 512];  // 16 A-frags (K=512) (16KB)
  __shared__ __bf16 lds_d   [16 * 64];   // 2 A-frags  (K=64)
  __shared__ __bf16 lds_cm  [16 * 128];  // 4 A-frags  [gamma_x, m]
  __shared__ __bf16 lds_xc  [16 * 64];   // 2 A-frags
  __shared__ __bf16 lds_inp [16 * 128];  // 4 A-frags  [c_c, m]
  __shared__ float  lds_xh  [16 * 64];
  __shared__ float  lds_zh  [16 * 64];

  const int tid  = threadIdx.x;
  const int wave = tid >> 5;
  const int lane = tid & 31;
  const int r0   = blockIdx.x * 16;   // batch-row tile
  const int ncsub = lane & 15;
  const int mbase = (lane >> 4) * 8;

  // stage 2 column owned by waves 0-3
  const int ncol2 = (wave & 3) * 16 + ncsub;

  // stage 2a pack constants (waves 4-7, thread owns column k of [gamma_x|m])
  const int kpack = tid - 128;  // valid when wave>=4: 0..127
  float wgx_d = 0.f, bgx_c = 0.f;
  if (wave >= 4 && kpack < 64) {
    wgx_d = Wgx[(size_t)kpack * Cc + kpack];
    bgx_c = b_gx[kpack];
  }

  // LSTM cell state lives entirely in registers (thread owns same 32 cells every step)
  float creg[4][8] = {};

  // init hidden state to zero
  for (int i = tid; i < 16 * 512; i += 256) lds_h[i] = (__bf16)0.f;
  __syncthreads();

#pragma unroll 1
  for (int t = 0; t < Tt; ++t) {
    // ---- stage 0: pack deltas rows into A-frags ----
#pragma unroll 1
    for (int i = tid; i < 16 * 64; i += 256) {
      int m = i >> 6, k = i & 63;
      float dv = deltas[((size_t)(r0 + m) * Tt + t) * Cc + k];
      lds_d[afrag_addr(m, k)] = (__bf16)dv;
    }
    __syncthreads();

    // ---- stage 1: gamma_h = exp(-relu(d @ Wgh^T + b)); h_dec = gamma_h * h_prev ----
#pragma unroll 1
    for (int it = 0; it < 4; ++it) {
      int nt = wave * 4 + it;
      int ncol = nt * 16 + ncsub;
      float bgv = b_gh[ncol];
      v8f acc = {};
#pragma unroll
      for (int kt = 0; kt < 2; ++kt)
        acc = wmma_bf16(load_afrag(lds_d, kt, lane), load_bfrag(Wgh_p, kt, nt, 32, lane), acc);
#pragma unroll
      for (int r = 0; r < 8; ++r) {
        int m = mbase + r;
        float g = __expf(-fmaxf(acc[r] + bgv, 0.f));
        float hd = g * (float)lds_h[m * 512 + ncol];
        lds_hdec[afrag_addr(m, ncol)] = (__bf16)hd;
      }
    }
    __syncthreads();

    // ---- stage 2a: waves 0-3: x_h GEMM (K=512); waves 4-7: pack [gamma_x|m] + m-part of inp ----
    if (wave >= 4) {
      int k = kpack;
#pragma unroll 2
      for (int m = 0; m < 16; ++m) {
        __bf16 v;
        if (k < 64) {
          float dv = deltas[((size_t)(r0 + m) * Tt + t) * Cc + k];
          v = (__bf16)__expf(-fmaxf(dv * wgx_d + bgx_c, 0.f));
        } else {
          float mv = masks[((size_t)(r0 + m) * Tt + t) * Cc + (k - 64)];
          v = (__bf16)mv;
          lds_inp[afrag_addr(m, k)] = v;
        }
        lds_cm[afrag_addr(m, k)] = v;
      }
    } else {
      v8f acc = {};
#pragma unroll 4
      for (int kt = 0; kt < 16; ++kt)
        acc = wmma_bf16(load_afrag(lds_hdec, kt, lane), load_bfrag(Whist_p, kt, wave, 4, lane), acc);
      float bhist_c = b_hist[ncol2];
#pragma unroll
      for (int r = 0; r < 8; ++r) {
        int m = mbase + r;
        size_t gidx = ((size_t)(r0 + m) * Tt + t) * Cc + ncol2;
        float xh = acc[r] + bhist_c;
        float mv = masks[gidx];
        float xv = values[gidx];
        float xc = mv * xv + (1.f - mv) * xh;
        lds_xh[m * 64 + ncol2] = xh;
        lds_xc[afrag_addr(m, ncol2)] = (__bf16)xc;
      }
    }
    __syncthreads();

    // ---- stage 2b: z_h = x_c @ Wfeat_m^T + b_feat ----
    if (wave < 4) {
      v8f acc = {};
#pragma unroll
      for (int kt = 0; kt < 2; ++kt)
        acc = wmma_bf16(load_afrag(lds_xc, kt, lane), load_bfrag(Wfeat_p, kt, wave, 4, lane), acc);
      float bfeat_c = b_feat[ncol2];
#pragma unroll
      for (int r = 0; r < 8; ++r)
        lds_zh[(mbase + r) * 64 + ncol2] = acc[r] + bfeat_c;
    }
    __syncthreads();

    // ---- stage 2c: alpha; c_h; c_c -> out, pack inp cols 0..63 ----
    if (wave < 4) {
      v8f acc = {};
#pragma unroll 2
      for (int kt = 0; kt < 4; ++kt)
        acc = wmma_bf16(load_afrag(lds_cm, kt, lane), load_bfrag(Wcomb_p, kt, wave, 4, lane), acc);
      float bcomb_c = b_comb[ncol2];
#pragma unroll
      for (int r = 0; r < 8; ++r) {
        int m = mbase + r;
        size_t gidx = ((size_t)(r0 + m) * Tt + t) * Cc + ncol2;
        float alpha = acc[r] + bcomb_c;
        float xh = lds_xh[m * 64 + ncol2];
        float zh = lds_zh[m * 64 + ncol2];
        float ch = alpha * zh + (1.f - alpha) * xh;
        float mv = masks[gidx];
        float xv = values[gidx];
        float cc = mv * xv + (1.f - mv) * ch;
        lds_inp[afrag_addr(m, ncol2)] = (__bf16)cc;
        out[gidx] = cc;
      }
    }
    __syncthreads();

    // ---- stage 3: gates = inp @ Wih^T + h_dec @ Whh^T + bias; LSTM update in registers ----
#pragma unroll
    for (int g = 0; g < 4; ++g) {
      const int ntg = g * 8 + wave;  // n-tile within H (0..31)
      v8f a0 = {}, a1 = {}, a2 = {}, a3 = {};
#pragma unroll 2
      for (int kt = 0; kt < 4; ++kt) {
        v16bf a = load_afrag(lds_inp, kt, lane);
        a0 = wmma_bf16(a, load_bfrag(Wih_p, kt, 0 * 32 + ntg, 128, lane), a0);
        a1 = wmma_bf16(a, load_bfrag(Wih_p, kt, 1 * 32 + ntg, 128, lane), a1);
        a2 = wmma_bf16(a, load_bfrag(Wih_p, kt, 2 * 32 + ntg, 128, lane), a2);
        a3 = wmma_bf16(a, load_bfrag(Wih_p, kt, 3 * 32 + ntg, 128, lane), a3);
      }
#pragma unroll 2
      for (int kt = 0; kt < 16; ++kt) {
        v16bf a = load_afrag(lds_hdec, kt, lane);
        a0 = wmma_bf16(a, load_bfrag(Whh_p, kt, 0 * 32 + ntg, 128, lane), a0);
        a1 = wmma_bf16(a, load_bfrag(Whh_p, kt, 1 * 32 + ntg, 128, lane), a1);
        a2 = wmma_bf16(a, load_bfrag(Whh_p, kt, 2 * 32 + ntg, 128, lane), a2);
        a3 = wmma_bf16(a, load_bfrag(Whh_p, kt, 3 * 32 + ntg, 128, lane), a3);
      }
      int ncol = ntg * 16 + ncsub;
      float bI = b_ih[0 * Hh + ncol] + b_hh[0 * Hh + ncol];
      float bF = b_ih[1 * Hh + ncol] + b_hh[1 * Hh + ncol];
      float bG = b_ih[2 * Hh + ncol] + b_hh[2 * Hh + ncol];
      float bO = b_ih[3 * Hh + ncol] + b_hh[3 * Hh + ncol];
#pragma unroll
      for (int r = 0; r < 8; ++r) {
        float gi = a0[r] + bI;
        float gf = a1[r] + bF;
        float gg = a2[r] + bG;
        float go = a3[r] + bO;
        float cn = sigm(gf) * creg[g][r] + sigm(gi) * tanhf(gg);
        creg[g][r] = cn;
        float hn = sigm(go) * tanhf(cn);
        lds_h[(mbase + r) * 512 + ncol] = (__bf16)hn;
      }
    }
    __syncthreads();  // h/hdec handoff into next step
  }
}

// ---------------- host ----------------
extern "C" void kernel_launch(void* const* d_in, const int* in_sizes, int n_in,
                              void* d_out, int out_size, void* d_ws, size_t ws_size,
                              hipStream_t stream) {
  const float* data   = (const float*)d_in[0];
  const float* W_ih   = (const float*)d_in[1];
  const float* W_hh   = (const float*)d_in[2];
  const float* b_ih   = (const float*)d_in[3];
  const float* b_hh   = (const float*)d_in[4];
  const float* W_gh   = (const float*)d_in[5];
  const float* b_gh   = (const float*)d_in[6];
  const float* W_gx   = (const float*)d_in[7];
  const float* b_gx   = (const float*)d_in[8];
  const float* W_hist = (const float*)d_in[9];
  const float* b_hist = (const float*)d_in[10];
  const float* W_feat = (const float*)d_in[11];
  const float* b_feat = (const float*)d_in[12];
  const float* W_comb = (const float*)d_in[13];
  const float* b_comb = (const float*)d_in[14];
  float* out = (float*)d_out;

  char* wsb = (char*)d_ws;
  size_t off = 0;
  auto alloc = [&](size_t bytes) -> void* {
    void* p = wsb + off;
    off += (bytes + 63) & ~(size_t)63;
    return p;
  };
  const size_t NBTC = (size_t)Bb * Tt * Cc;
  float* values = (float*)alloc(NBTC * 4);
  float* masks  = (float*)alloc(NBTC * 4);
  float* deltas = (float*)alloc(NBTC * 4);
  __bf16* Wgh_p   = (__bf16*)alloc((size_t)64 * 512 * 2);
  __bf16* Whist_p = (__bf16*)alloc((size_t)512 * 64 * 2);
  __bf16* Wfeat_p = (__bf16*)alloc((size_t)64 * 64 * 2);
  __bf16* Wcomb_p = (__bf16*)alloc((size_t)128 * 64 * 2);
  __bf16* Wih_p   = (__bf16*)alloc((size_t)128 * 2048 * 2);
  __bf16* Whh_p   = (__bf16*)alloc((size_t)512 * 2048 * 2);

  prep_kernel<<<(Bb * Cc + 255) / 256, 256, 0, stream>>>(data, values, masks, deltas);

  pack_kernel<<<256, 256, 0, stream>>>(W_gh,   Wgh_p,   512,  64,  0);
  pack_kernel<<<256, 256, 0, stream>>>(W_hist, Whist_p, 64,   512, 0);
  pack_kernel<<<64,  256, 0, stream>>>(W_feat, Wfeat_p, 64,   64,  1);
  pack_kernel<<<64,  256, 0, stream>>>(W_comb, Wcomb_p, 64,   128, 0);
  pack_kernel<<<512, 256, 0, stream>>>(W_ih,   Wih_p,   2048, 128, 0);
  pack_kernel<<<512, 256, 0, stream>>>(W_hh,   Whh_p,   2048, 512, 0);

  scan_kernel<<<Bb / 16, 256, 0, stream>>>(
      values, masks, deltas,
      Wgh_p, b_gh, Whist_p, b_hist, Wfeat_p, b_feat, Wcomb_p, b_comb,
      W_gx, b_gx, Wih_p, b_ih, Whh_p, b_hh, out);
}
